// Attention_16698832847178
// MI455X (gfx1250) — matrix operand order repaired and checked by reading, not compiled, not run on hardware
//
#include <hip/hip_runtime.h>
#include <math.h>

// ---------------------------------------------------------------------------
// MI455X (gfx1250) fused adapter-attention. All heavy math via
// V_WMMA_F32_16X16X32_F16 (wave32). Weights & V pre-transposed so every
// global->LDS staging is a contiguous 16B GLOBAL_LOAD_ASYNC_TO_LDS_B128
// (ASYNCcnt), double-buffered so prefetch of tile i+1 overlaps WMMA on tile i
// (async loads complete in order -> s_wait_asynccnt <= #prefetch is safe).
// Fragment layouts per CDNA5 ISA 7.12.2.
// ---------------------------------------------------------------------------

typedef __attribute__((ext_vector_type(16))) _Float16 v16h;
typedef __attribute__((ext_vector_type(8)))  _Float16 v8h;
typedef __attribute__((ext_vector_type(8)))  float    v8f;
typedef __attribute__((ext_vector_type(4)))  int      v4i;
typedef __attribute__((address_space(1))) v4i gv4i;   // global int4
typedef __attribute__((address_space(3))) v4i lv4i;   // LDS int4

#define BB  2
#define SS  2048
#define DDm 2048
#define HH  16
#define HDD 128
#define LLa 10
#define Mtot 4096      // B*S

#if defined(__HIP_DEVICE_COMPILE__) && __has_builtin(__builtin_amdgcn_global_load_async_to_lds_b128)
#define HAVE_ASYNC_LDS 1
#else
#define HAVE_ASYNC_LDS 0
#endif

// 16-byte global -> LDS staging (async on CDNA5 when available)
static __device__ __forceinline__ void stage16(const _Float16* g, _Float16* l) {
#if HAVE_ASYNC_LDS
  // addrspace(1) ptr is 64-bit (flat global VA == global VA);
  // addrspace(3) ptr is 32-bit (low 32 bits of flat LDS addr == LDS offset).
  __builtin_amdgcn_global_load_async_to_lds_b128(
      (gv4i*)(unsigned long long)(size_t)g,
      (lv4i*)(unsigned)(size_t)l, 0, 0);
#else
  *(v8h*)l = *(const v8h*)g;
#endif
}
// Wait until at most `n` async-to-LDS instructions remain outstanding.
template <int N>
static __device__ __forceinline__ void wait_async() {
#if HAVE_ASYNC_LDS
#if __has_builtin(__builtin_amdgcn_s_wait_asynccnt)
  __builtin_amdgcn_s_wait_asynccnt(N);
#else
  asm volatile("s_wait_asynccnt %0" :: "i"(N) : "memory");
#endif
#endif
}

static __device__ __forceinline__ v16h frag_join(v8h lo, v8h hi) {
  return __builtin_shufflevector(lo, hi, 0,1,2,3,4,5,6,7,8,9,10,11,12,13,14,15);
}
// A-style fragment: 8 halves at p, 8 halves at p+16
static __device__ __forceinline__ v16h ld_frag_gap(const _Float16* p) {
  v8h lo = *(const v8h*)p;
  v8h hi = *(const v8h*)(p + 16);
  return frag_join(lo, hi);
}
// B-style fragment: 8 halves at p, 8 halves at p+16
static __device__ __forceinline__ v16h ld_frag_seq(const _Float16* p) {
  v8h lo = *(const v8h*)p;
  v8h hi = *(const v8h*)(p + 16);
  return frag_join(lo, hi);
}
static __device__ __forceinline__ v8f vzero8() {
  v8f z;
#pragma unroll
  for (int i = 0; i < 8; ++i) z[i] = 0.0f;
  return z;
}

// ---------------------------------------------------------------------------
__global__ __launch_bounds__(256) void cvt_f32_f16(const float* __restrict__ in,
                                                   _Float16* __restrict__ out, int n) {
  for (int i = blockIdx.x * 256 + threadIdx.x; i < n; i += gridDim.x * 256)
    out[i] = (_Float16)in[i];
}

// in: [R][C] f32 row-major -> out: [C][R] f16 (weights: K-major for GEMM B side)
__global__ __launch_bounds__(256) void transpose_f32_f16(const float* __restrict__ in,
                                                         _Float16* __restrict__ out,
                                                         int R, int C) {
  __shared__ float t[32][33];
  const int r0 = blockIdx.y * 32, c0 = blockIdx.x * 32;
  const int tx = threadIdx.x & 31, ty = threadIdx.x >> 5;
#pragma unroll
  for (int i = 0; i < 32; i += 8)
    t[ty + i][tx] = in[(size_t)(r0 + ty + i) * C + c0 + tx];
  __syncthreads();
#pragma unroll
  for (int i = 0; i < 32; i += 8)
    out[(size_t)(c0 + ty + i) * R + r0 + tx] = (_Float16)t[tx][ty + i];
}

// V (B,S,H,HD) f32 -> vt[(b*H+h)*HD + d][S] f16  (key-contiguous for PV B-frags)
__global__ __launch_bounds__(256) void transpose_v_f16(const float* __restrict__ v,
                                                       _Float16* __restrict__ vt) {
  __shared__ float t[32][33];
  const int bh = blockIdx.z;
  const int b = bh >> 4, h = bh & 15;
  const int s0 = blockIdx.x * 32, d0 = blockIdx.y * 32;
  const int tx = threadIdx.x & 31, ty = threadIdx.x >> 5;
#pragma unroll
  for (int i = 0; i < 32; i += 8)
    t[ty + i][tx] = v[(size_t)(b * SS + s0 + ty + i) * DDm + h * HDD + d0 + tx];
  __syncthreads();
#pragma unroll
  for (int i = 0; i < 32; i += 8)
    vt[((size_t)bh * HDD + d0 + ty + i) * SS + s0 + tx] = (_Float16)t[tx][ty + i];
}

// ---------------------------------------------------------------------------
// C[4096,2048](f32) = A[4096,2048](f16, row-major) * Bt[2048,2048](f16, [n][k]).
// Block 128x128, 8 waves (4x2) of 32x64. Double-buffered async LDS, K-step 32.
__global__ __launch_bounds__(256) void gemm_f16_wmma(const _Float16* __restrict__ A,
                                                     const _Float16* __restrict__ Bt,
                                                     float* __restrict__ C) {
  __shared__ _Float16 sA[2][128][40];   // 2 x 10 KB
  __shared__ _Float16 sB[2][128][40];   // 2 x 10 KB
  const int tid = threadIdx.x, lane = tid & 31, wid = tid >> 5;
  const int wm = wid & 3, wn = wid >> 2;
  const int bm = blockIdx.x * 128, bn = blockIdx.y * 128;
  const int n  = lane & 15;
  const int mb = (lane >> 4) * 8, kg = (lane >> 4) * 8, kh = (lane >> 4) * 16;

  auto stage_tiles = [&](int buf, int k0) {
#pragma unroll
    for (int it = 0; it < 2; ++it) {
      int c = tid + it * 256;           // 0..511
      int r = c >> 2, co = (c & 3) * 8; // 4 chunks of 8 halves per 32-wide row
      stage16(A  + (size_t)(bm + r) * DDm + k0 + co, &sA[buf][r][co]);
      stage16(Bt + (size_t)(bn + r) * DDm + k0 + co, &sB[buf][r][co]);
    }
  };

  v8f acc[2][4];
#pragma unroll
  for (int i = 0; i < 2; ++i)
#pragma unroll
    for (int j = 0; j < 4; ++j) acc[i][j] = vzero8();

  stage_tiles(0, 0);                    // prologue prefetch (4 async insts/wave)
  for (int k0 = 0; k0 < DDm; k0 += 32) {
    const int cur = (k0 >> 5) & 1, nxt = cur ^ 1;
    if (k0 + 32 < DDm) {                // prefetch next tile under this compute
      stage_tiles(nxt, k0 + 32);
      wait_async<4>();                  // only current tile's loads must land
    } else {
      wait_async<0>();
    }
    __syncthreads();

    v16h af[2], bf[4];
#pragma unroll
    for (int i = 0; i < 2; ++i)
      af[i] = ld_frag_gap(&sA[cur][wm * 32 + i * 16 + n][kg]);
#pragma unroll
    for (int j = 0; j < 4; ++j)
      bf[j] = ld_frag_seq(&sB[cur][wn * 64 + j * 16 + n][kg]);
#pragma unroll
    for (int i = 0; i < 2; ++i)
#pragma unroll
      for (int j = 0; j < 4; ++j)
        acc[i][j] = __builtin_amdgcn_wmma_f32_16x16x32_f16(
            false, af[i], false, bf[j], (short)0, acc[i][j], false, false);
    __syncthreads();                    // all reads of `cur` done before restage
  }
  float* cb = C + (size_t)(bm + wm * 32 + mb) * DDm + (bn + wn * 64 + n);
#pragma unroll
  for (int i = 0; i < 2; ++i)
#pragma unroll
    for (int j = 0; j < 4; ++j)
#pragma unroll
      for (int r = 0; r < 8; ++r)
        cb[(size_t)(i * 16 + r) * DDm + j * 16] = acc[i][j][r];
}

// ---------------------------------------------------------------------------
// RoPE q,k in place (f32) and emit f16 copies. One thread per (b,s,h,d/2).
__global__ __launch_bounds__(256) void rope_cvt(float* __restrict__ q, float* __restrict__ k,
                                                const float* __restrict__ cs,
                                                const float* __restrict__ sn,
                                                _Float16* __restrict__ qh,
                                                _Float16* __restrict__ kh) {
  unsigned idx = blockIdx.x * 256 + threadIdx.x;    // B*S*H*64 = 2^22
  int d2 = idx & 63;
  int h  = (idx >> 6) & 15;
  int s  = (idx >> 10) & 2047;
  int b  = idx >> 21;
  size_t base = ((size_t)(b * SS + s)) * DDm + h * HDD + 2 * d2;
  float c = cs[s * 64 + d2], si = sn[s * 64 + d2];
  float q0 = q[base], q1 = q[base + 1];
  float r0 = q0 * c - q1 * si, r1 = q0 * si + q1 * c;
  q[base] = r0; q[base + 1] = r1;
  qh[base] = (_Float16)r0; qh[base + 1] = (_Float16)r1;
  float k0 = k[base], k1 = k[base + 1];
  float t0 = k0 * c - k1 * si, t1 = k0 * si + k1 * c;
  k[base] = t0; k[base + 1] = t1;
  kh[base] = (_Float16)t0; kh[base + 1] = (_Float16)t1;
}

// ---------------------------------------------------------------------------
// Causal flash attention. Block = 4 waves x 16 q rows. grid = (B*H, S/64).
// Double-buffered async K/V staging.
__global__ __launch_bounds__(128) void flash_attn_wmma(const _Float16* __restrict__ qg,
                                                       const _Float16* __restrict__ kg_,
                                                       const _Float16* __restrict__ vt,
                                                       float* __restrict__ o, float scale) {
  __shared__ _Float16 sK[2][32][136];     // [key][d], +8 pad
  __shared__ _Float16 sVt[2][128][40];    // [d][key], +8 pad
  __shared__ _Float16 sP[4][16][40];      // per-wave P staging
  const int bhid = blockIdx.x;
  const int b = bhid >> 4, h = bhid & 15;
  const int q0b = blockIdx.y * 64;
  const int tid = threadIdx.x, lane = tid & 31, wid = tid >> 5;
  const int q0  = q0b + wid * 16;
  const int n   = lane & 15;
  const int mb  = (lane >> 4) * 8, kg = (lane >> 4) * 8, kh = (lane >> 4) * 16;

  auto stage_kv = [&](int buf, int kt) {
#pragma unroll
    for (int it = 0; it < 4; ++it) {
      int c  = tid + it * 128;            // 0..511
      int kr = c >> 4, co = (c & 15) * 8;
      stage16(kg_ + (size_t)(b * SS + kt + kr) * DDm + h * HDD + co, &sK[buf][kr][co]);
      int d = c >> 2, ko = (c & 3) * 8;
      stage16(vt + ((size_t)bhid * HDD + d) * SS + kt + ko, &sVt[buf][d][ko]);
    }
  };

  v16h qa[4];
  {
    const _Float16* qp = qg + ((size_t)(b * SS + q0 + n) * DDm + h * HDD);
#pragma unroll
    for (int kc = 0; kc < 4; ++kc) {
      v8h lo = *(const v8h*)(qp + kc * 32 + kg);
      v8h hi = *(const v8h*)(qp + kc * 32 + 16 + kg);
      qa[kc] = frag_join(lo, hi);
    }
  }
  v8f acc[8];
#pragma unroll
  for (int nc = 0; nc < 8; ++nc) acc[nc] = vzero8();
  float mrow[8], lrow[8];
#pragma unroll
  for (int j = 0; j < 8; ++j) { mrow[j] = -1e30f; lrow[j] = 0.0f; }

  const int kend = q0b + 64;
  stage_kv(0, 0);                         // prologue prefetch (8 async insts/wave)
  for (int kt = 0; kt < kend; kt += 32) {
    const int cur = (kt >> 5) & 1, nxt = cur ^ 1;
    if (kt + 32 < kend) {
      stage_kv(nxt, kt + 32);
      wait_async<8>();
    } else {
      wait_async<0>();
    }
    __syncthreads();

    // Scores: two 16-key halves, HD=128 in 4 WMMA k-slices each
    v8f sf[2];
#pragma unroll
    for (int nh = 0; nh < 2; ++nh) {
      v8f s = vzero8();
#pragma unroll
      for (int kc = 0; kc < 4; ++kc) {
        v16h bfr = ld_frag_seq(&sK[cur][nh * 16 + n][kc * 32 + kg]);
        s = __builtin_amdgcn_wmma_f32_16x16x32_f16(false, qa[kc], false, bfr,
                                                   (short)0, s, false, false);
      }
      sf[nh] = s;
    }

    // Online softmax
#pragma unroll
    for (int j = 0; j < 8; ++j) {
      int   qi = q0 + mb + j;
      float s0 = sf[0][j] * scale + ((kt + n)      > qi ? -1e9f : 0.0f);
      float s1 = sf[1][j] * scale + ((kt + 16 + n) > qi ? -1e9f : 0.0f);
      float mx = fmaxf(s0, s1);
#pragma unroll
      for (int m = 1; m < 16; m <<= 1) mx = fmaxf(mx, __shfl_xor(mx, m, 32));
      float mnew  = fmaxf(mrow[j], mx);
      float alpha = __expf(mrow[j] - mnew);
      mrow[j] = mnew;
      float p0 = __expf(s0 - mnew), p1 = __expf(s1 - mnew);
      float rs = p0 + p1;
#pragma unroll
      for (int m = 1; m < 16; m <<= 1) rs += __shfl_xor(rs, m, 32);
      lrow[j] = lrow[j] * alpha + rs;
#pragma unroll
      for (int nc = 0; nc < 8; ++nc) acc[nc][j] *= alpha;
      sP[wid][mb + j][n]      = (_Float16)p0;
      sP[wid][mb + j][16 + n] = (_Float16)p1;
    }

    // O += P(16x32) * V(32x128)
    v16h pa = ld_frag_gap(&sP[wid][n][kg]);
#pragma unroll
    for (int nc = 0; nc < 8; ++nc) {
      v16h bfr = ld_frag_seq(&sVt[cur][nc * 16 + n][kg]);
      acc[nc] = __builtin_amdgcn_wmma_f32_16x16x32_f16(false, pa, false, bfr,
                                                       (short)0, acc[nc], false, false);
    }
    __syncthreads();                      // all reads of `cur` done before restage
  }

  float* ob = o + (size_t)(b * SS + q0 + mb) * DDm + h * HDD + n;
#pragma unroll
  for (int nc = 0; nc < 8; ++nc)
#pragma unroll
    for (int j = 0; j < 8; ++j)
      ob[(size_t)j * DDm + nc * 16] = acc[nc][j] / lrow[j];
}

// ---------------------------------------------------------------------------
__global__ __launch_bounds__(256) void adapter_kv(const float* __restrict__ adp,
                                                  const float* __restrict__ wk,
                                                  const float* __restrict__ wv,
                                                  float* __restrict__ ak,
                                                  float* __restrict__ av) {
  int idx = blockIdx.x * 256 + threadIdx.x;
  if (idx >= LLa * DDm) return;
  int kk = idx >> 11;
  int nn = idx & 2047;
  float s1 = 0.0f, s2 = 0.0f;
  for (int d = 0; d < DDm; ++d) {
    float a = adp[kk * DDm + d];
    s1 += a * wk[(size_t)d * DDm + nn];
    s2 += a * wv[(size_t)d * DDm + nn];
  }
  ak[idx] = s1;
  av[idx] = s2;
}

__global__ __launch_bounds__(256) void adapter_attn(const float* __restrict__ q,
                                                    const float* __restrict__ ak,
                                                    const float* __restrict__ av,
                                                    const float* __restrict__ gate,
                                                    float* __restrict__ o, float scale) {
  __shared__ float sak[LLa][HDD];
  __shared__ float sav[LLa][HDD];
  const int h = blockIdx.y;
  for (int i = threadIdx.x; i < LLa * HDD; i += 256) {
    int kk = i >> 7, dd = i & 127;
    sak[kk][dd] = ak[(size_t)(kk * HH + h) * HDD + dd];
    sav[kk][dd] = av[(size_t)(kk * HH + h) * HDD + dd];
  }
  __syncthreads();
  const int    rs   = blockIdx.x * 256 + threadIdx.x;
  const size_t base = (size_t)rs * DDm + h * HDD;
  float sc[LLa];
  float mx = -1e30f;
#pragma unroll
  for (int kk = 0; kk < LLa; ++kk) {
    float d = 0.0f;
    for (int dd = 0; dd < HDD; ++dd) d += q[base + dd] * sak[kk][dd];
    d *= scale;
    sc[kk] = d;
    mx = fmaxf(mx, d);
  }
  float sum = 0.0f;
#pragma unroll
  for (int kk = 0; kk < LLa; ++kk) { sc[kk] = __expf(sc[kk] - mx); sum += sc[kk]; }
  const float g = gate[h] / sum;
  for (int dd = 0; dd < HDD; ++dd) {
    float a = 0.0f;
#pragma unroll
    for (int kk = 0; kk < LLa; ++kk) a += sc[kk] * sav[kk][dd];
    o[base + dd] += g * a;
  }
}

// ---------------------------------------------------------------------------
extern "C" void kernel_launch(void* const* d_in, const int* in_sizes, int n_in,
                              void* d_out, int out_size, void* d_ws, size_t ws_size,
                              hipStream_t stream) {
  (void)in_sizes; (void)n_in; (void)out_size; (void)ws_size;
  const float* x    = (const float*)d_in[0];
  const float* wq   = (const float*)d_in[1];
  const float* wk   = (const float*)d_in[2];
  const float* wv   = (const float*)d_in[3];
  const float* wo   = (const float*)d_in[4];
  const float* adp  = (const float*)d_in[5];
  const float* gate = (const float*)d_in[6];
  const float* fc   = (const float*)d_in[7];
  const float* fs   = (const float*)d_in[8];
  float* out = (float*)d_out;

  const size_t ND = (size_t)Mtot * DDm;
  const size_t WD = (size_t)DDm * DDm;

  char* w = (char*)d_ws;
  auto bump = [&](size_t bytes) {
    char* p = w;
    w += (bytes + 255) & ~(size_t)255;
    return p;
  };
  _Float16* xb    = (_Float16*)bump(ND * 2);
  _Float16* wqt   = (_Float16*)bump(WD * 2);
  _Float16* wkt   = (_Float16*)bump(WD * 2);
  _Float16* wvt   = (_Float16*)bump(WD * 2);
  _Float16* wot   = (_Float16*)bump(WD * 2);
  float*    qf    = (float*)   bump(ND * 4);
  float*    kf    = (float*)   bump(ND * 4);
  float*    vf    = (float*)   bump(ND * 4);
  _Float16* qh    = (_Float16*)bump(ND * 2);
  _Float16* khh   = (_Float16*)bump(ND * 2);
  _Float16* vt    = (_Float16*)bump(ND * 2);
  float*    attnf = (float*)   bump(ND * 4);
  _Float16* attnh = (_Float16*)bump(ND * 2);
  float*    akf   = (float*)   bump((size_t)LLa * DDm * 4);
  float*    avf   = (float*)   bump((size_t)LLa * DDm * 4);

  const float scale = 0.08838834764831845f;  // 1/sqrt(128)

  cvt_f32_f16<<<2048, 256, 0, stream>>>(x, xb, (int)ND);
  dim3 tg(DDm / 32, DDm / 32);
  transpose_f32_f16<<<tg, 256, 0, stream>>>(wq, wqt, DDm, DDm);
  transpose_f32_f16<<<tg, 256, 0, stream>>>(wk, wkt, DDm, DDm);
  transpose_f32_f16<<<tg, 256, 0, stream>>>(wv, wvt, DDm, DDm);
  transpose_f32_f16<<<tg, 256, 0, stream>>>(wo, wot, DDm, DDm);

  dim3 gg(Mtot / 128, DDm / 128);   // (32, 16)
  gemm_f16_wmma<<<gg, 256, 0, stream>>>(xb, wqt, qf);
  gemm_f16_wmma<<<gg, 256, 0, stream>>>(xb, wkt, kf);
  gemm_f16_wmma<<<gg, 256, 0, stream>>>(xb, wvt, vf);

  rope_cvt<<<(BB * SS * HH * 64) / 256, 256, 0, stream>>>(qf, kf, fc, fs, qh, khh);
  transpose_v_f16<<<dim3(SS / 32, HDD / 32, BB * HH), 256, 0, stream>>>(vf, vt);

  adapter_kv<<<(LLa * DDm + 255) / 256, 256, 0, stream>>>(adp, wk, wv, akf, avf);

  flash_attn_wmma<<<dim3(BB * HH, SS / 64), 128, 0, stream>>>(qh, khh, vt, attnf, scale);

  adapter_attn<<<dim3(Mtot / 256, HH), 256, 0, stream>>>(qf, akf, avf, gate, attnf, scale);

  cvt_f32_f16<<<2048, 256, 0, stream>>>(attnf, attnh, (int)ND);
  gemm_f16_wmma<<<gg, 256, 0, stream>>>(attnh, wot, out);
}
